// MultiOmicGATModule_66907000537157
// MI455X (gfx1250) — matrix operand order repaired and checked by reading, not compiled, not run on hardware
//
#include <hip/hip_runtime.h>
#include <stdint.h>

// MultiOmic GATv2 encoder for MI455X (gfx1250, wave32).
// GEMMs: v_wmma_f32_16x16x32_bf16, 4 WMMAs/wave/K-step (32x32 wave tile,
// 128x64 block tile), double-buffered async global->LDS staging so the
// tensor DMA for tile k+1 overlaps the WMMAs of tile k (ASYNCcnt<=3 wait).
// Last K-step peeled so the steady-state loop is branch-free.
// Edge softmax: ordered-int atomicMax + float atomicAdd. LN: wave32 shuffles.

#define HID 256
#define HEADS 8
#define CH 32
#define BATCH 64
#define NEG_SLOPE 0.2f
#define LN_EPS 1e-5f

#define BM 128
#define BN 64
#define BK 32

typedef __attribute__((ext_vector_type(16))) __bf16 v16bf;
typedef __attribute__((ext_vector_type(8)))  float v8f;
typedef __attribute__((ext_vector_type(4)))  int   v4i;

#if __has_builtin(__builtin_amdgcn_global_load_async_to_lds_b128)
#define HAVE_ASYNC_LDS 1
#else
#define HAVE_ASYNC_LDS 0
#endif

template <int N>
__device__ __forceinline__ void async_wait() {
#if __has_builtin(__builtin_amdgcn_s_wait_asynccnt)
  __builtin_amdgcn_s_wait_asynccnt(N);
#elif defined(__gfx1250__)
  asm volatile("s_wait_asynccnt %0" ::"i"(N) : "memory");
#endif
}

__device__ __forceinline__ void copy16_g2l(const unsigned short* __restrict__ g,
                                           unsigned short* l) {
#if HAVE_ASYNC_LDS
  __builtin_amdgcn_global_load_async_to_lds_b128(
      (__attribute__((address_space(1))) v4i*)g,
      (__attribute__((address_space(3))) v4i*)l, 0, 0);
#else
  *(uint4*)(void*)l = *(const uint4*)(const void*)g;
#endif
}

__device__ __forceinline__ unsigned short f2bf(float f) {
  unsigned u = __float_as_uint(f);
  u += 0x7fffu + ((u >> 16) & 1u);   // round-to-nearest-even
  return (unsigned short)(u >> 16);
}
__device__ __forceinline__ unsigned f2ord(float f) {
  unsigned u = __float_as_uint(f);
  return (u & 0x80000000u) ? ~u : (u | 0x80000000u);
}
__device__ __forceinline__ float ord2f(unsigned u) {
  unsigned v = (u & 0x80000000u) ? (u & 0x7fffffffu) : ~u;
  return __uint_as_float(v);
}

// ---------------- elementwise / conversion ----------------

__global__ void k_zero(float* __restrict__ p, size_t n) {
  size_t i = (size_t)blockIdx.x * blockDim.x + threadIdx.x;
  for (; i < n; i += (size_t)gridDim.x * blockDim.x) p[i] = 0.f;
}

__global__ void k_f32_to_bf16(const float* __restrict__ in,
                              unsigned short* __restrict__ out, size_t n) {
  size_t i = (size_t)blockIdx.x * blockDim.x + threadIdx.x;
  for (; i < n; i += (size_t)gridDim.x * blockDim.x) out[i] = f2bf(in[i]);
}

// Wt(Nc x K) = transpose(W(K x Nc)) in bf16; coalesced writes.
__global__ void k_f32_to_bf16_T(const float* __restrict__ W,
                                unsigned short* __restrict__ Wt,
                                int K, int Nc) {
  int o = blockIdx.x * blockDim.x + threadIdx.x;
  if (o >= K * Nc) return;
  int n = o / K, k = o - n * K;
  Wt[o] = f2bf(W[(size_t)k * Nc + n]);
}

// A = x.T in bf16: x is (BATCH, N) row-major; A is (N, BATCH) row-major.
__global__ void k_transpose_bf16(const float* __restrict__ x,
                                 unsigned short* __restrict__ A, int N) {
  int i = blockIdx.x * blockDim.x + threadIdx.x;
  if (i >= N * BATCH) return;
  int n = i >> 6, b = i & 63;
  A[i] = f2bf(x[(size_t)b * N + n]);
}

// --------- WMMA GEMM: C(MxNc) = A(MxK)bf16 @ Wt(NcxK)bf16^T + bias --------
// Block 256 thr = 8 waves as 4(M) x 2(N); wave tile 32x32 = 4 WMMA tiles.
// Double-buffered LDS; steady-state loop stages tile k+1 (async DMA) then
// waits ASYNCcnt<=3 so tile k's data is resident while k+1 is in flight.

__global__ __launch_bounds__(256) void k_gemm_bias_wmma(
    const unsigned short* __restrict__ A,
    const unsigned short* __restrict__ Wt,
    const float* __restrict__ bias, float* __restrict__ C,
    int M, int K, int Nc) {
  __shared__ unsigned short Alds[2][BM * BK];   // [m][k] row-major, 2x8KB
  __shared__ unsigned short Blds[2][BN * BK];   // [n][k] row-major, 2x4KB

  const int tid  = threadIdx.x;
  const int wave = tid >> 5;
  const int lane = tid & 31;
  const int wm   = wave >> 1;             // 0..3 : 32-row wave tile
  const int wn   = wave & 1;              // 0..1 : 32-col wave tile
  const int tiles_n = Nc / BN;
  const int bm = blockIdx.x / tiles_n;
  const int bn = blockIdx.x % tiles_n;
  const int row0 = bm * BM;
  const int col0 = bn * BN;
  const int half = lane >> 4;
  const int l15  = lane & 15;

  // per-thread staging coordinates (3 x 16B per thread per tile)
  int ar0 = (tid >> 2), ak0 = (tid & 3) * 8;            // A chunk 0
  int ar1 = ((tid + 256) >> 2), ak1 = ak0;              // A chunk 1
  int bn0 = (tid >> 2), bk0 = (tid & 3) * 8;            // B chunk
  int garow0 = row0 + ar0; garow0 = garow0 < M ? garow0 : (M - 1);
  int garow1 = row0 + ar1; garow1 = garow1 < M ? garow1 : (M - 1);

  auto stage = [&](int buf, int k0) {
    copy16_g2l(A + (size_t)garow0 * K + k0 + ak0, &Alds[buf][ar0 * BK + ak0]);
    copy16_g2l(A + (size_t)garow1 * K + k0 + ak1, &Alds[buf][ar1 * BK + ak1]);
    copy16_g2l(Wt + (size_t)(col0 + bn0) * K + k0 + bk0,
               &Blds[buf][bn0 * BK + bk0]);
  };

  v8f acc[2][2] = {};

  auto compute = [&](int cur) {
    union Frag { v16bf v; uint4 q[2]; };
    Frag af[2], bfr[2];
#pragma unroll
    for (int mi = 0; mi < 2; ++mi) {
      // A frag 16x32: lane m=l15, K chunks [half*8,+7] and [16+half*8,+7]
      const unsigned short* ap = &Alds[cur][(wm * 32 + mi * 16 + l15) * BK];
      af[mi].q[0] = *(const uint4*)(const void*)(ap + half * 8);
      af[mi].q[1] = *(const uint4*)(const void*)(ap + 16 + half * 8);
    }
#pragma unroll
    for (int ni = 0; ni < 2; ++ni) {
      // B frag 32x16: lane n=l15, K chunks [half*8,+7] and [16+half*8,+7]
      const unsigned short* bp =
          &Blds[cur][(wn * 32 + ni * 16 + l15) * BK];
      bfr[ni].q[0] = *(const uint4*)(const void*)(bp + half * 8);
      bfr[ni].q[1] = *(const uint4*)(const void*)(bp + 16 + half * 8);
    }
#pragma unroll
    for (int mi = 0; mi < 2; ++mi)
#pragma unroll
      for (int ni = 0; ni < 2; ++ni)
        acc[mi][ni] = __builtin_amdgcn_wmma_f32_16x16x32_bf16(
            false, af[mi].v, false, bfr[ni].v, (short)0, acc[mi][ni],
            false, false);
  };

  const int ksteps = K / BK;
  stage(0, 0);
  // steady state: branch-free except the loop counter
  for (int ks = 0; ks < ksteps - 1; ++ks) {
    stage((ks + 1) & 1, (ks + 1) * BK);  // DMA for next tile in flight
    async_wait<3>();                     // retire current tile's 3 loads
    __syncthreads();
    compute(ks & 1);
    __syncthreads();                     // readers done before re-stage
  }
  // peeled last tile
  async_wait<0>();
  __syncthreads();
  compute((ksteps - 1) & 1);

  // C/D layout: lane half selects M rows 0-7 vs 8-15 across 8 VGPRs
#pragma unroll
  for (int mi = 0; mi < 2; ++mi)
#pragma unroll
    for (int ni = 0; ni < 2; ++ni) {
      int col   = col0 + wn * 32 + ni * 16 + l15;
      int rbase = row0 + wm * 32 + mi * 16 + half * 8;
      float bv  = bias ? bias[col] : 0.f;
#pragma unroll
      for (int r = 0; r < 8; ++r) {
        int row = rbase + r;
        if (row < M) C[(size_t)row * Nc + col] = acc[mi][ni][r] + bv;
      }
    }
}

// ---------------- LayerNorm (wave per row, wave32 shuffles) ---------------
// mode 0: out = relu(LN(src; g,b))
// mode 1: out = LN( elu(src + abias) + resid ; g,b)   (in-place over resid ok)

__global__ __launch_bounds__(256) void k_norm(
    const float* __restrict__ src, const float* __restrict__ resid,
    const float* __restrict__ abias, const float* __restrict__ g,
    const float* __restrict__ b, float* __restrict__ out, int N, int mode) {
  int wave = threadIdx.x >> 5, lane = threadIdx.x & 31;
  int row = blockIdx.x * 8 + wave;
  if (row >= N) return;
  const float* sp = src + (size_t)row * HID;
  float v[8];
  float s = 0.f;
#pragma unroll
  for (int j = 0; j < 8; ++j) {
    int c = lane + 32 * j;
    float x = sp[c];
    if (mode == 1) {
      x += abias[c];
      x = x > 0.f ? x : (expf(x) - 1.f);           // ELU
      x += resid[(size_t)row * HID + c];
    }
    v[j] = x; s += x;
  }
#pragma unroll
  for (int o = 16; o; o >>= 1) s += __shfl_xor(s, o);
  float mean = s * (1.f / HID);
  float var = 0.f;
#pragma unroll
  for (int j = 0; j < 8; ++j) { float d = v[j] - mean; var += d * d; }
#pragma unroll
  for (int o = 16; o; o >>= 1) var += __shfl_xor(var, o);
  float rstd = rsqrtf(var * (1.f / HID) + LN_EPS);
  float* op = out + (size_t)row * HID;
#pragma unroll
  for (int j = 0; j < 8; ++j) {
    int c = lane + 32 * j;
    float y = (v[j] - mean) * rstd * g[c] + b[c];
    if (mode == 0) y = y > 0.f ? y : 0.f;          // ReLU
    op[c] = y;
  }
}

// ---------------- GATv2 edge kernels (self loops synthesized) -------------

__device__ __forceinline__ void edge_sd(const int* ei, int E, int e,
                                        int& s, int& d) {
  if (e < E) { s = ei[e]; d = ei[E + e]; }
  else       { s = d = e - E; }
}

__global__ void k_edge_score(const float* __restrict__ xl,
                             const float* __restrict__ xr,
                             const int* __restrict__ ei,
                             const float* __restrict__ att,
                             float* __restrict__ score,
                             unsigned* __restrict__ nodemax, int E, int ET) {
  int i = blockIdx.x * blockDim.x + threadIdx.x;
  if (i >= ET * HEADS) return;
  int e = i >> 3, hh = i & 7;
  int s, d; edge_sd(ei, E, e, s, d);
  const float4* lp = (const float4*)(xl + (size_t)s * HID + hh * CH);
  const float4* rp = (const float4*)(xr + (size_t)d * HID + hh * CH);
  const float4* ap = (const float4*)(att + hh * CH);
  float acc = 0.f;
#pragma unroll
  for (int c = 0; c < CH / 4; ++c) {
    float4 l = lp[c], r = rp[c], a = ap[c];
    float t;
    t = l.x + r.x; t = t > 0.f ? t : NEG_SLOPE * t; acc += t * a.x;
    t = l.y + r.y; t = t > 0.f ? t : NEG_SLOPE * t; acc += t * a.y;
    t = l.z + r.z; t = t > 0.f ? t : NEG_SLOPE * t; acc += t * a.z;
    t = l.w + r.w; t = t > 0.f ? t : NEG_SLOPE * t; acc += t * a.w;
  }
  score[i] = acc;
  atomicMax(&nodemax[d * HEADS + hh], f2ord(acc));
}

__global__ void k_edge_exp(const int* __restrict__ ei,
                           float* __restrict__ score,
                           const unsigned* __restrict__ nodemax,
                           float* __restrict__ den, int E, int ET) {
  int i = blockIdx.x * blockDim.x + threadIdx.x;
  if (i >= ET * HEADS) return;
  int e = i >> 3, hh = i & 7;
  int s, d; edge_sd(ei, E, e, s, d);
  float m = ord2f(nodemax[d * HEADS + hh]);
  float ex = expf(score[i] - m);
  score[i] = ex;
  atomicAdd(&den[d * HEADS + hh], ex);
}

__global__ void k_edge_agg(const float* __restrict__ xl,
                           const int* __restrict__ ei,
                           const float* __restrict__ score,
                           const float* __restrict__ den,
                           float* __restrict__ agg, int E, int ET) {
  int i = blockIdx.x * blockDim.x + threadIdx.x;
  if (i >= ET * HEADS) return;
  int e = i >> 3, hh = i & 7;
  int s, d; edge_sd(ei, E, e, s, d);
  float alpha = score[i] / den[d * HEADS + hh];
  const float4* lp = (const float4*)(xl + (size_t)s * HID + hh * CH);
  float* op = agg + (size_t)d * HID + hh * CH;
#pragma unroll
  for (int c = 0; c < CH / 4; ++c) {
    float4 l = lp[c];
    atomicAdd(&op[4 * c + 0], alpha * l.x);
    atomicAdd(&op[4 * c + 1], alpha * l.y);
    atomicAdd(&op[4 * c + 2], alpha * l.z);
    atomicAdd(&op[4 * c + 3], alpha * l.w);
  }
}

// ---------------- pooling / output ----------------------------------------

__global__ void k_pool(const float* __restrict__ h, float* __restrict__ z,
                       size_t n) {
  size_t i = (size_t)blockIdx.x * blockDim.x + threadIdx.x;
  for (; i < n; i += (size_t)gridDim.x * blockDim.x)
    atomicAdd(&z[i & (HID - 1)], h[i]);
}

__global__ void k_write_out(const float* __restrict__ z,
                            float* __restrict__ out, float invN) {
  int i = blockIdx.x * blockDim.x + threadIdx.x;
  if (i >= BATCH * HID) return;
  out[i] = z[i & (HID - 1)] * invN;
}

// ---------------- host orchestration ---------------------------------------

namespace {

struct Ws {
  unsigned short *A, *W, *hbf;
  float *h, *xl, *xr;            // xr buffer reused as agg output
  float *score;
  unsigned *nodemax;
  float *den, *z;
};

inline int gs(size_t n, int bs = 256) {
  size_t g = (n + bs - 1) / bs;
  return (int)(g > 65535u * 16u ? 65535u * 16u : (g ? g : 1));
}

void gemm(const unsigned short* A, const unsigned short* Wt, const float* bias,
          float* C, int M, int K, int Nc, hipStream_t st) {
  int grid = ((M + BM - 1) / BM) * (Nc / BN);
  k_gemm_bias_wmma<<<grid, 256, 0, st>>>(A, Wt, bias, C, M, K, Nc);
}

void run_encoder(const float* x, const int* ei, const float* const* P,
                 int N, int E, float* out_slice, const Ws& w, hipStream_t st) {
  const int ET = E + N;
  // input projection: h = relu(LN(x.T @ W_in + b_in))
  k_transpose_bf16<<<gs((size_t)N * BATCH), 256, 0, st>>>(x, w.A, N);
  k_f32_to_bf16_T<<<gs((size_t)BATCH * HID), 256, 0, st>>>(P[0], w.W,
                                                           BATCH, HID);
  gemm(w.A, w.W, P[1], w.h, N, BATCH, HID, st);
  k_norm<<<(N + 7) / 8, 256, 0, st>>>(w.h, nullptr, nullptr, P[2], P[3],
                                      w.h, N, 0);
  for (int layer = 0; layer < 2; ++layer) {
    const float* const* L = P + 4 + 8 * layer;  // Wl,bl,Wr,br,att,bias,g,b
    k_f32_to_bf16<<<gs((size_t)N * HID), 256, 0, st>>>(w.h, w.hbf,
                                                       (size_t)N * HID);
    k_f32_to_bf16_T<<<gs((size_t)HID * HID), 256, 0, st>>>(L[0], w.W,
                                                           HID, HID);
    gemm(w.hbf, w.W, L[1], w.xl, N, HID, HID, st);
    k_f32_to_bf16_T<<<gs((size_t)HID * HID), 256, 0, st>>>(L[2], w.W,
                                                           HID, HID);
    gemm(w.hbf, w.W, L[3], w.xr, N, HID, HID, st);

    k_zero<<<gs((size_t)N * HEADS), 256, 0, st>>>((float*)w.nodemax,
                                                  (size_t)N * HEADS);
    k_zero<<<gs((size_t)N * HEADS), 256, 0, st>>>(w.den, (size_t)N * HEADS);
    k_edge_score<<<gs((size_t)ET * HEADS), 256, 0, st>>>(
        w.xl, w.xr, ei, L[4], w.score, w.nodemax, E, ET);
    k_edge_exp<<<gs((size_t)ET * HEADS), 256, 0, st>>>(ei, w.score, w.nodemax,
                                                       w.den, E, ET);
    // xr no longer needed -> reuse as aggregation output
    k_zero<<<gs((size_t)N * HID), 256, 0, st>>>(w.xr, (size_t)N * HID);
    k_edge_agg<<<gs((size_t)ET * HEADS), 256, 0, st>>>(w.xl, ei, w.score,
                                                       w.den, w.xr, E, ET);
    // h = LN(elu(agg + bias) + h)
    k_norm<<<(N + 7) / 8, 256, 0, st>>>(w.xr, w.h, L[5], L[6], L[7],
                                        w.h, N, 1);
  }
  k_zero<<<1, 256, 0, st>>>(w.z, HID);
  k_pool<<<gs((size_t)N * HID), 256, 0, st>>>(w.h, w.z, (size_t)N * HID);
  k_write_out<<<(BATCH * HID + 255) / 256, 256, 0, st>>>(w.z, out_slice,
                                                         1.f / (float)N);
}

}  // namespace

extern "C" void kernel_launch(void* const* d_in, const int* in_sizes, int n_in,
                              void* d_out, int out_size, void* d_ws,
                              size_t ws_size, hipStream_t stream) {
  (void)in_sizes; (void)n_in; (void)out_size; (void)ws_size;

  const float* gene_x  = (const float*)d_in[0];
  const float* meth_x  = (const float*)d_in[1];
  const float* mirna_x = (const float*)d_in[2];
  const int*   gene_ei  = (const int*)d_in[3];
  const int*   meth_ei  = (const int*)d_in[4];
  const int*   mirna_ei = (const int*)d_in[5];

  // params flattened in dict insertion order: per encoder
  // W_in,b_in,g_in,be_in, then 2x {Wl,bl,Wr,br,att,bias,ln_g,ln_b}
  const float* P[3][20];
  for (int m = 0; m < 3; ++m)
    for (int j = 0; j < 20; ++j)
      P[m][j] = (const float*)d_in[6 + m * 20 + j];

  const int NMAX = 20000, ETMAX = 320000 + 20000;
  char* p = (char*)d_ws;
  auto carve = [&](size_t bytes) -> void* {
    void* r = (void*)p;
    p += (bytes + 255) & ~(size_t)255;
    return r;
  };
  Ws w;
  w.A       = (unsigned short*)carve((size_t)NMAX * BATCH * 2);
  w.W       = (unsigned short*)carve((size_t)HID * HID * 2);
  w.hbf     = (unsigned short*)carve((size_t)NMAX * HID * 2);
  w.h       = (float*)carve((size_t)NMAX * HID * 4);
  w.xl      = (float*)carve((size_t)NMAX * HID * 4);
  w.xr      = (float*)carve((size_t)NMAX * HID * 4);
  w.score   = (float*)carve((size_t)ETMAX * HEADS * 4);
  w.nodemax = (unsigned*)carve((size_t)NMAX * HEADS * 4);
  w.den     = (float*)carve((size_t)NMAX * HEADS * 4);
  w.z       = (float*)carve((size_t)HID * 4);

  float* out = (float*)d_out;
  run_encoder(gene_x,  gene_ei,  P[0], 10000, 160000, out,                   w, stream);
  run_encoder(meth_x,  meth_ei,  P[1], 20000, 320000, out + BATCH * HID,     w, stream);
  run_encoder(mirna_x, mirna_ei, P[2], 2000,  32000,  out + 2 * BATCH * HID, w, stream);
}
